// DfNet_88665304859180
// MI455X (gfx1250) — matrix operand order repaired and checked by reading, not compiled, not run on hardware
//
#include <hip/hip_runtime.h>
#include <math.h>

typedef __attribute__((ext_vector_type(16))) _Float16 v16h;
typedef __attribute__((ext_vector_type(8)))  _Float16 v8h;
typedef __attribute__((ext_vector_type(8)))  float    v8f;

#define B_ 16
#define T_ 1000
#define F_ 481
#define NBERB 32
#define NBDF 96
#define BT_ (B_*T_)

__device__ __forceinline__ float sigm(float x){ return 1.0f/(1.0f + expf(-x)); }

// ---------------------------------------------------------------- f32 -> f16
__global__ void k_cvt16(const float* __restrict__ s, _Float16* __restrict__ d, int n){
  int i = blockIdx.x*blockDim.x + threadIdx.x;
  if (i < n) d[i] = (_Float16)s[i];
}

// ---------------------------------------------------------------- WMMA GEMM
// Y[M,N] = act( X[M,K] * W[N,K]^T + bias ).  One wave per 16x32 output tile
// (two adjacent 16x16 N-tiles sharing the A fragment).
__global__ void k_gemm(const _Float16* __restrict__ X, const _Float16* __restrict__ W,
                       const float* __restrict__ bias, float* __restrict__ Y,
                       int M, int N, int K, int act){
  int wave = blockIdx.x*(blockDim.x >> 5) + (threadIdx.x >> 5);
  int lane = threadIdx.x & 31;
  int nt2 = N >> 5;
  int tm = wave / nt2, tn2 = wave - tm*nt2;
  if (tm*16 >= M) return;
  int g = lane >> 4, l = lane & 15;
  int nb = tn2*32;
  v8f acc0 = {}, acc1 = {};
  const _Float16* xrow = X + (size_t)(tm*16 + l)*K;
  const _Float16* w0 = W + (size_t)(nb + l)*K + 16*g;
  const _Float16* w1 = w0 + (size_t)16*K;
  for (int k0 = 0; k0 < K; k0 += 32){
    union { v16h v; v8h h[2]; } a;
    a.h[0] = *(const v8h*)(xrow + k0 + 16*g);
    a.h[1] = *(const v8h*)(xrow + k0 + 16*g + 8);
    v16h b0 = *(const v16h*)(w0 + k0);
    v16h b1 = *(const v16h*)(w1 + k0);
    acc0 = __builtin_amdgcn_wmma_f32_16x16x32_f16(false, a.v, false, b0, (short)0, acc0, false, false);
    acc1 = __builtin_amdgcn_wmma_f32_16x16x32_f16(false, a.v, false, b1, (short)0, acc1, false, false);
  }
  #pragma unroll
  for (int v = 0; v < 8; ++v){
    int m = tm*16 + v + 8*g;
    float y0 = acc0[v], y1 = acc1[v];
    int n0 = nb + l, n1 = nb + 16 + l;
    if (bias){ y0 += bias[n0]; y1 += bias[n1]; }
    if (act == 1){ y0 = fmaxf(y0, 0.0f); y1 = fmaxf(y1, 0.0f); }
    else if (act == 2){ y0 = tanhf(y0); y1 = tanhf(y1); }
    Y[(size_t)m*N + n0] = y0;
    Y[(size_t)m*N + n1] = y1;
  }
}

// ------------------------------------------------------- GRU recurrence (WMMA)
// 768 threads = 24 waves; H=256, 3H=768, batch=16 (one WMMA M tile).
// Each wave owns a 16x32 slice of gh, sharing the A (hidden-state) fragment.
__device__ __forceinline__ void gru_body(const float* __restrict__ gi,
                                         const _Float16* __restrict__ Whh,
                                         const float* __restrict__ bih,
                                         const float* __restrict__ bhh,
                                         float* __restrict__ ys,
                                         _Float16* hbuf, float* gh){
  const int tid = threadIdx.x;
  const int wv = tid >> 5, lane = tid & 31;
  const int g = lane >> 4, l = lane & 15;
  for (int i = tid; i < 16*256; i += 768) hbuf[i] = (_Float16)0.0f;
  float hreg[6];
  #pragma unroll
  for (int s = 0; s < 6; ++s) hreg[s] = 0.0f;
  __syncthreads();

  const int tn0 = wv*2;
  const _Float16* w0 = Whh + (size_t)(tn0*16 + l)*256 + 16*g;
  const _Float16* w1 = w0 + (size_t)16*256;
  const _Float16* hrow = hbuf + l*256;

  for (int t = 0; t < T_; ++t){
    v8f acc0 = {}, acc1 = {};
    #pragma unroll
    for (int k0 = 0; k0 < 256; k0 += 32){
      union { v16h v; v8h h[2]; } a;
      a.h[0] = *(const v8h*)(hrow + k0 + 16*g);
      a.h[1] = *(const v8h*)(hrow + k0 + 16*g + 8);
      v16h b0 = *(const v16h*)(w0 + k0);
      v16h b1 = *(const v16h*)(w1 + k0);
      acc0 = __builtin_amdgcn_wmma_f32_16x16x32_f16(false, a.v, false, b0, (short)0, acc0, false, false);
      acc1 = __builtin_amdgcn_wmma_f32_16x16x32_f16(false, a.v, false, b1, (short)0, acc1, false, false);
    }
    #pragma unroll
    for (int v = 0; v < 8; ++v){
      gh[(v + 8*g)*768 + tn0*16 + l]      = acc0[v];
      gh[(v + 8*g)*768 + tn0*16 + 16 + l] = acc1[v];
    }
    __syncthreads();
    int s = 0;
    for (int idx = tid; idx < 16*256; idx += 768, ++s){
      int b = idx >> 8, j = idx & 255;
      const float* gib = gi + ((size_t)(b*T_ + t))*768;
      float r = sigm(gib[j]       + bih[j]     + gh[b*768 + j]       + bhh[j]);
      float z = sigm(gib[256 + j] + bih[256+j] + gh[b*768 + 256 + j] + bhh[256+j]);
      float n = tanhf(gib[512 + j] + bih[512+j] + r*(gh[b*768 + 512 + j] + bhh[512+j]));
      float h = (1.0f - z)*n + z*hreg[s];
      hreg[s] = h;
      ys[((size_t)(b*T_ + t))*256 + j] = h;
      hbuf[b*256 + j] = (_Float16)h;
    }
    __syncthreads();
  }
}

__global__ __launch_bounds__(768) void k_gru(const float* __restrict__ gi,
                                             const _Float16* __restrict__ Whh,
                                             const float* __restrict__ bih,
                                             const float* __restrict__ bhh,
                                             float* __restrict__ ys){
  __shared__ _Float16 hbuf[16*256];
  __shared__ float    gh[16*768];
  gru_body(gi, Whh, bih, bhh, ys, hbuf, gh);
}

// Two independent GRUs in one launch (blockIdx selects parameter set) — the
// erb-decoder GRU and df-GRU#0 both depend only on emb, so run concurrently.
__global__ __launch_bounds__(768) void k_gru2(const float* __restrict__ gi0,
                                              const _Float16* __restrict__ W0,
                                              const float* __restrict__ bih0,
                                              const float* __restrict__ bhh0,
                                              float* __restrict__ ys0,
                                              const float* __restrict__ gi1,
                                              const _Float16* __restrict__ W1,
                                              const float* __restrict__ bih1,
                                              const float* __restrict__ bhh1,
                                              float* __restrict__ ys1){
  __shared__ _Float16 hbuf[16*256];
  __shared__ float    gh[16*768];
  if (blockIdx.x == 0) gru_body(gi0, W0, bih0, bhh0, ys0, hbuf, gh);
  else                 gru_body(gi1, W1, bih1, bhh1, ys1, hbuf, gh);
}

// ------------------------------------------------ encoder conv: erb_conv0 (3x3)
__global__ void k_conv_erb0(const float* __restrict__ x, const float* __restrict__ dw,
                            const float* __restrict__ gg, const float* __restrict__ bb,
                            float* __restrict__ y){
  int idx = blockIdx.x*blockDim.x + threadIdx.x;
  if (idx >= B_*16*T_*NBERB) return;
  int f = idx % NBERB; int t = (idx/NBERB) % T_;
  int c = (idx/(NBERB*T_)) % 16; int b = idx/(NBERB*T_*16);
  float acc = 0.0f;
  for (int kt = 0; kt < 3; ++kt){
    int ti = t - 2 + kt; if (ti < 0) continue;
    for (int kf = 0; kf < 3; ++kf){
      int fi = f - 1 + kf; if (fi < 0 || fi >= NBERB) continue;
      acc += x[((size_t)(b*T_ + ti))*NBERB + fi] * dw[c*9 + kt*3 + kf];
    }
  }
  y[idx] = fmaxf(acc*gg[c] + bb[c], 0.0f);
}

// ------------------------------------------------ df_conv0 (3x3, groups=2 + pw)
__global__ void k_conv_df0(const float* __restrict__ fs, const float* __restrict__ dw,
                           const float* __restrict__ pw, const float* __restrict__ gg,
                           const float* __restrict__ bb, float* __restrict__ y){
  int idx = blockIdx.x*blockDim.x + threadIdx.x;
  if (idx >= B_*T_*NBDF) return;
  int f = idx % NBDF; int t = (idx/NBDF) % T_; int b = idx/(NBDF*T_);
  float dwt[16];
  for (int co = 0; co < 16; ++co){
    int ci = co >> 3;
    float a = 0.0f;
    for (int kt = 0; kt < 3; ++kt){
      int ti = t - 2 + kt; if (ti < 0) continue;
      for (int kf = 0; kf < 3; ++kf){
        int fi = f - 1 + kf; if (fi < 0 || fi >= NBDF) continue;
        a += fs[(((size_t)(b*T_ + ti))*NBDF + fi)*2 + ci] * dw[co*9 + kt*3 + kf];
      }
    }
    dwt[co] = a;
  }
  for (int co = 0; co < 16; ++co){
    float a = 0.0f;
    for (int ci = 0; ci < 16; ++ci) a += pw[co*16+ci]*dwt[ci];
    y[(((size_t)(b*16) + co)*T_ + t)*NBDF + f] = fmaxf(a*gg[co] + bb[co], 0.0f);
  }
}

// ------------------- depthwise kf=3 (stride / lhs-dilated "transpose") + pw 16x16
__global__ void k_dwpw(const float* __restrict__ x, const float* __restrict__ dw,
                       const float* __restrict__ pw, const float* __restrict__ gg,
                       const float* __restrict__ bb, float* __restrict__ y,
                       int Fin, int Fout, int stride, int transpose){
  int idx = blockIdx.x*blockDim.x + threadIdx.x;
  if (idx >= B_*T_*Fout) return;
  int f = idx % Fout; int t = (idx/Fout) % T_; int b = idx/(Fout*T_);
  float dwt[16];
  for (int c = 0; c < 16; ++c){
    float a = 0.0f;
    const float* xc = x + ((size_t)(b*16 + c)*T_ + t)*Fin;
    if (!transpose){
      for (int kf = 0; kf < 3; ++kf){
        int fi = f*stride - 1 + kf;
        if (fi >= 0 && fi < Fin) a += xc[fi]*dw[c*3+kf];
      }
    } else {
      for (int kf = 0; kf < 3; ++kf){
        int di = f + kf - 1;          // index into lhs-dilated(2), left-pad 1
        if (di >= 0 && di <= 2*Fin - 2 && (di & 1) == 0) a += xc[di>>1]*dw[c*3+kf];
      }
    }
    dwt[c] = a;
  }
  for (int co = 0; co < 16; ++co){
    float a = 0.0f;
    for (int ci = 0; ci < 16; ++ci) a += pw[co*16+ci]*dwt[ci];
    y[((size_t)(b*16 + co)*T_ + t)*Fout + f] = fmaxf(a*gg[co] + bb[co], 0.0f);
  }
}

// ---------------------------------------- 1x1 depthwise conv block + skip add
__global__ void k_pw1_add(const float* __restrict__ x, const float* __restrict__ dw,
                          const float* __restrict__ gg, const float* __restrict__ bb,
                          const float* __restrict__ add, float* __restrict__ y, int F){
  int idx = blockIdx.x*blockDim.x + threadIdx.x;
  if (idx >= B_*16*T_*F) return;
  int c = (idx/(F*T_)) % 16;
  float a = x[idx]*dw[c]*gg[c] + bb[c];
  y[idx] = fmaxf(a, 0.0f) + add[idx];
}

// ------------------------------------------- conv0_out: 16->1, kf=3, sigmoid
__global__ void k_conv_out(const float* __restrict__ x, const float* __restrict__ dw,
                           const float* __restrict__ gg, const float* __restrict__ bb,
                           float* __restrict__ m){
  int idx = blockIdx.x*blockDim.x + threadIdx.x;
  if (idx >= B_*T_*NBERB) return;
  int f = idx % NBERB; int t = (idx/NBERB) % T_; int b = idx/(NBERB*T_);
  float a = 0.0f;
  for (int ci = 0; ci < 16; ++ci)
    for (int kf = 0; kf < 3; ++kf){
      int fi = f - 1 + kf; if (fi < 0 || fi >= NBERB) continue;
      a += x[((size_t)(b*16 + ci)*T_ + t)*NBERB + fi] * dw[ci*3 + kf];
    }
  m[idx] = sigm(a*gg[0] + bb[0]);
}

// -------------------------------------- pack c1 [B,16,T,48] -> f16 [BT,768]
__global__ void k_pack_c1(const float* __restrict__ c1, _Float16* __restrict__ o){
  int idx = blockIdx.x*blockDim.x + threadIdx.x;
  if (idx >= BT_*768) return;
  int j = idx % 768; int bt = idx/768; int b = bt/T_, t = bt - b*T_;
  int f = j >> 4, c = j & 15;
  o[idx] = (_Float16)c1[((size_t)(b*16 + c)*T_ + t)*48 + f];
}

// ---------------- emb_in = pack(e3) + cemb (in-place f32) + f16 copy
__global__ void k_emb_in(const float* __restrict__ e3, float* __restrict__ cemb,
                         _Float16* __restrict__ o16){
  int idx = blockIdx.x*blockDim.x + threadIdx.x;
  if (idx >= BT_*128) return;
  int j = idx % 128; int bt = idx/128; int b = bt/T_, t = bt - b*T_;
  int f = j >> 4, c = j & 15;
  float v = e3[((size_t)(b*16 + c)*T_ + t)*8 + f] + cemb[idx];
  cemb[idx] = v;
  o16[idx] = (_Float16)v;
}

// ---------------- unpack d [BT,128] -> image [B,16,T,8]
__global__ void k_unpack_d(const float* __restrict__ dl, float* __restrict__ di){
  int idx = blockIdx.x*blockDim.x + threadIdx.x;
  if (idx >= B_*16*T_*8) return;
  int f = idx % 8; int t = (idx/8) % T_; int c = (idx/(8*T_)) % 16; int b = idx/(8*T_*16);
  di[idx] = dl[((size_t)(b*T_ + t))*128 + f*16 + c];
}

// ---------------- row head: out[r] = sigmoid(dot(X[r,:256],W)+b)*scale+off
__global__ void k_rowdot(const float* __restrict__ X, const float* __restrict__ W,
                         const float* __restrict__ bias, float* __restrict__ o,
                         float scale, float off){
  int r = blockIdx.x*blockDim.x + threadIdx.x;
  if (r >= BT_) return;
  float a = bias[0];
  const float* xr = X + (size_t)r*256;
  for (int j = 0; j < 256; ++j) a += xr[j]*W[j];
  o[r] = sigm(a)*scale + off;
}

// ---------------- final: mask einsum, deep-filter, alpha blend -> spec_out
__global__ void k_final(const float* __restrict__ spec, const float* __restrict__ erb_fb,
                        const float* __restrict__ m, const float* __restrict__ coefs,
                        const float* __restrict__ c0, const float* __restrict__ dwp,
                        const float* __restrict__ gp, const float* __restrict__ bp,
                        const float* __restrict__ alpha, float* __restrict__ out){
  int idx = blockIdx.x*blockDim.x + threadIdx.x;
  if (idx >= B_*T_*F_) return;
  int f = idx % F_; int t = (idx/F_) % T_; int b = idx/(F_*T_);
  const float* mr = m + ((size_t)(b*T_ + t))*NBERB;
  float mask = 0.0f;
  for (int e = 0; e < NBERB; ++e) mask += mr[e]*erb_fb[e*F_ + f];
  size_t sidx = (((size_t)(b*T_) + t)*F_ + f)*2;
  float smr = spec[sidx]*mask, smi = spec[sidx+1]*mask;
  float orr = smr, oii = smi;
  if (f < NBDF){
    float c0p[10];
    for (int o = 0; o < 10; ++o){
      int gs = (o < 5) ? 0 : 8;
      float a = 0.0f;
      for (int il = 0; il < 8; ++il)
        a += dwp[o*8 + il] * c0[((size_t)(b*16 + gs + il)*T_ + t)*NBDF + f];
      c0p[o] = fmaxf(a*gp[o] + bp[o], 0.0f);
    }
    const float* cf = coefs + ((size_t)(b*T_ + t))*960 + f*10;
    float dr = 0.0f, di = 0.0f;
    for (int k = 0; k < 5; ++k){
      float cr = cf[2*k]   + c0p[2*k];
      float ci = cf[2*k+1] + c0p[2*k+1];
      int ts = t + k - 4;
      float sr = 0.0f, si = 0.0f;
      if (ts >= 0){
        size_t s2 = (((size_t)(b*T_) + ts)*F_ + f)*2;
        sr = spec[s2]; si = spec[s2+1];
      }
      dr += cr*sr - ci*si;
      di += cr*si + ci*sr;
    }
    float a = alpha[(size_t)b*T_ + t];
    orr = a*dr + (1.0f - a)*smr;
    oii = a*di + (1.0f - a)*smi;
  }
  out[sidx] = orr; out[sidx+1] = oii;
}

// ================================================================ host side
static inline int cdiv(long n, int b){ return (int)((n + b - 1)/b); }

extern "C" void kernel_launch(void* const* d_in, const int* in_sizes, int n_in,
                              void* d_out, int out_size, void* d_ws, size_t ws_size,
                              hipStream_t stream){
  (void)in_sizes; (void)n_in; (void)out_size; (void)ws_size;
  #define IN(i) ((const float*)d_in[i])
  char* ws = (char*)d_ws;
  size_t off = 0;
  #define ALLOCF(name, nelem) float* name = (float*)(ws + off); off += (((size_t)(nelem)*4 + 255) & ~(size_t)255)
  #define ALLOCH(name, nelem) _Float16* name = (_Float16*)(ws + off); off += (((size_t)(nelem)*2 + 255) & ~(size_t)255)

  // f16 weights
  ALLOCH(w_dffc, 128*768);  ALLOCH(w_ih0, 768*128);  ALLOCH(w_hh0, 768*256);
  ALLOCH(w_ihd, 768*256);   ALLOCH(w_hhd, 768*256);  ALLOCH(w_fce, 128*256);
  ALLOCH(w_ihf0, 768*256);  ALLOCH(w_hhf0, 768*256);
  ALLOCH(w_ihf1, 768*256);  ALLOCH(w_hhf1, 768*256);
  ALLOCH(w_ihf2, 768*256);  ALLOCH(w_hhf2, 768*256);
  ALLOCH(w_dfo, 960*256);
  // activations
  ALLOCF(e0, (size_t)B_*16*T_*32);  ALLOCF(e1, (size_t)B_*16*T_*16);
  ALLOCF(e2, (size_t)B_*16*T_*8);   ALLOCF(e3, (size_t)B_*16*T_*8);
  ALLOCF(c0, (size_t)B_*16*T_*96);
  ALLOCF(c1gi, (size_t)BT_*768);            // shared: c1 [B,16,T,48] then gi [BT,768]
  ALLOCF(gi2, (size_t)BT_*768);             // second gi (parallel GRU pair)
  ALLOCH(c1h, (size_t)BT_*768);
  ALLOCF(cemb, (size_t)BT_*128);            // becomes emb_in
  ALLOCH(embin16, (size_t)BT_*128);
  ALLOCF(emb, (size_t)BT_*256);  ALLOCH(emb16, (size_t)BT_*256);
  ALLOCF(dbuf, (size_t)BT_*256); ALLOCH(dbuf16, (size_t)BT_*256);
  ALLOCF(dlin, (size_t)BT_*128); ALLOCF(dimg, (size_t)B_*16*T_*8);
  ALLOCF(P, (size_t)B_*16*T_*32); ALLOCF(Q, (size_t)B_*16*T_*32);
  ALLOCF(h1, (size_t)BT_*256); ALLOCF(h2, (size_t)BT_*256); ALLOCF(h3, (size_t)BT_*256);
  ALLOCH(hf16a, (size_t)BT_*256); ALLOCH(hf16b, (size_t)BT_*256); ALLOCH(hf16c, (size_t)BT_*256);
  ALLOCF(coefs, (size_t)BT_*960);

  float* out_spec  = (float*)d_out;
  float* out_m     = out_spec + (size_t)B_*T_*F_*2;
  float* out_lsnr  = out_m + (size_t)B_*T_*NBERB;
  float* out_alpha = out_lsnr + BT_;

  const int TB = 256;
  // ---- weight conversions
  k_cvt16<<<cdiv(128*768,TB),TB,0,stream>>>(IN(27), w_dffc, 128*768);
  k_cvt16<<<cdiv(768*128,TB),TB,0,stream>>>(IN(29), w_ih0, 768*128);
  k_cvt16<<<cdiv(768*256,TB),TB,0,stream>>>(IN(30), w_hh0, 768*256);
  k_cvt16<<<cdiv(768*256,TB),TB,0,stream>>>(IN(35), w_ihd, 768*256);
  k_cvt16<<<cdiv(768*256,TB),TB,0,stream>>>(IN(36), w_hhd, 768*256);
  k_cvt16<<<cdiv(128*256,TB),TB,0,stream>>>(IN(39), w_fce, 128*256);
  k_cvt16<<<cdiv(768*256,TB),TB,0,stream>>>(IN(68), w_ihf0, 768*256);
  k_cvt16<<<cdiv(768*256,TB),TB,0,stream>>>(IN(69), w_hhf0, 768*256);
  k_cvt16<<<cdiv(768*256,TB),TB,0,stream>>>(IN(72), w_ihf1, 768*256);
  k_cvt16<<<cdiv(768*256,TB),TB,0,stream>>>(IN(73), w_hhf1, 768*256);
  k_cvt16<<<cdiv(768*256,TB),TB,0,stream>>>(IN(76), w_ihf2, 768*256);
  k_cvt16<<<cdiv(768*256,TB),TB,0,stream>>>(IN(77), w_hhf2, 768*256);
  k_cvt16<<<cdiv(960*256,TB),TB,0,stream>>>(IN(83), w_dfo, 960*256);

  // ---- encoder
  k_conv_erb0<<<cdiv((size_t)B_*16*T_*32,TB),TB,0,stream>>>(IN(1), IN(4), IN(5), IN(6), e0);
  k_dwpw<<<cdiv((size_t)BT_*16,TB),TB,0,stream>>>(e0, IN(7), IN(10), IN(8), IN(9), e1, 32,16,2,0);
  k_dwpw<<<cdiv((size_t)BT_*8,TB),TB,0,stream>>>(e1, IN(11), IN(14), IN(12), IN(13), e2, 16,8,2,0);
  k_dwpw<<<cdiv((size_t)BT_*8,TB),TB,0,stream>>>(e2, IN(15), IN(18), IN(16), IN(17), e3, 8,8,1,0);
  k_conv_df0<<<cdiv((size_t)BT_*96,TB),TB,0,stream>>>(IN(2), IN(19), IN(22), IN(20), IN(21), c0);
  k_dwpw<<<cdiv((size_t)BT_*48,TB),TB,0,stream>>>(c0, IN(23), IN(26), IN(24), IN(25), c1gi, 96,48,2,0);
  k_pack_c1<<<cdiv((size_t)BT_*768,TB),TB,0,stream>>>(c1gi, c1h);
  k_gemm<<<cdiv((BT_/16)*(128/32),8), 256,0,stream>>>(c1h, w_dffc, IN(28), cemb, BT_,128,768,0);
  k_emb_in<<<cdiv((size_t)BT_*128,TB),TB,0,stream>>>(e3, cemb, embin16);

  // ---- emb GRU
  k_gemm<<<cdiv((BT_/16)*(768/32),8), 256,0,stream>>>(embin16, w_ih0, (const float*)nullptr, c1gi, BT_,768,128,0);
  k_gru<<<1,768,0,stream>>>(c1gi, w_hh0, IN(31), IN(32), emb);
  k_cvt16<<<cdiv((size_t)BT_*256,TB),TB,0,stream>>>(emb, emb16, BT_*256);
  k_rowdot<<<cdiv(BT_,TB),TB,0,stream>>>(emb, IN(33), IN(34), out_lsnr, 50.0f, -15.0f);

  // ---- erb-dec GRU and df-GRU#0 run concurrently (both depend only on emb)
  k_gemm<<<cdiv((BT_/16)*(768/32),8), 256,0,stream>>>(emb16, w_ihd,  (const float*)nullptr, c1gi, BT_,768,256,0);
  k_gemm<<<cdiv((BT_/16)*(768/32),8), 256,0,stream>>>(emb16, w_ihf0, (const float*)nullptr, gi2,  BT_,768,256,0);
  k_gru2<<<2,768,0,stream>>>(c1gi, w_hhd, IN(37), IN(38), dbuf,
                             gi2,  w_hhf0, IN(70), IN(71), h1);

  // ---- erb decoder fc + convs (ping-pong P/Q)
  k_cvt16<<<cdiv((size_t)BT_*256,TB),TB,0,stream>>>(dbuf, dbuf16, BT_*256);
  k_gemm<<<cdiv((BT_/16)*(128/32),8), 256,0,stream>>>(dbuf16, w_fce, IN(40), dlin, BT_,128,256,1);
  k_unpack_d<<<cdiv((size_t)B_*16*T_*8,TB),TB,0,stream>>>(dlin, dimg);
  k_pw1_add<<<cdiv((size_t)B_*16*T_*8,TB),TB,0,stream>>>(e3, IN(41), IN(42), IN(43), dimg, P, 8);
  k_dwpw<<<cdiv((size_t)BT_*8,TB),TB,0,stream>>>(P, IN(44), IN(47), IN(45), IN(46), Q, 8,8,1,0);     // x3
  k_pw1_add<<<cdiv((size_t)B_*16*T_*8,TB),TB,0,stream>>>(e2, IN(48), IN(49), IN(50), Q, P, 8);
  k_dwpw<<<cdiv((size_t)BT_*16,TB),TB,0,stream>>>(P, IN(51), IN(54), IN(52), IN(53), Q, 8,16,1,1);   // x2 (transposed)
  k_pw1_add<<<cdiv((size_t)B_*16*T_*16,TB),TB,0,stream>>>(e1, IN(55), IN(56), IN(57), Q, P, 16);
  k_dwpw<<<cdiv((size_t)BT_*32,TB),TB,0,stream>>>(P, IN(58), IN(61), IN(59), IN(60), Q, 16,32,1,1);  // x1 (transposed)
  k_pw1_add<<<cdiv((size_t)B_*16*T_*32,TB),TB,0,stream>>>(e0, IN(62), IN(63), IN(64), Q, P, 32);
  k_conv_out<<<cdiv((size_t)BT_*32,TB),TB,0,stream>>>(P, IN(65), IN(66), IN(67), out_m);

  // ---- remaining df decoder GRU chain
  k_cvt16<<<cdiv((size_t)BT_*256,TB),TB,0,stream>>>(h1, hf16a, BT_*256);
  k_gemm<<<cdiv((BT_/16)*(768/32),8), 256,0,stream>>>(hf16a, w_ihf1, (const float*)nullptr, c1gi, BT_,768,256,0);
  k_gru<<<1,768,0,stream>>>(c1gi, w_hhf1, IN(74), IN(75), h2);
  k_cvt16<<<cdiv((size_t)BT_*256,TB),TB,0,stream>>>(h2, hf16b, BT_*256);
  k_gemm<<<cdiv((BT_/16)*(768/32),8), 256,0,stream>>>(hf16b, w_ihf2, (const float*)nullptr, c1gi, BT_,768,256,0);
  k_gru<<<1,768,0,stream>>>(c1gi, w_hhf2, IN(78), IN(79), h3);
  k_cvt16<<<cdiv((size_t)BT_*256,TB),TB,0,stream>>>(h3, hf16c, BT_*256);

  // ---- heads + deep filtering
  k_rowdot<<<cdiv(BT_,TB),TB,0,stream>>>(h3, IN(85), IN(86), out_alpha, 1.0f, 0.0f);
  k_gemm<<<cdiv((BT_/16)*(960/32),8), 256,0,stream>>>(hf16c, w_dfo, IN(84), coefs, BT_,960,256,2);
  k_final<<<cdiv((size_t)B_*T_*F_,TB),TB,0,stream>>>(IN(0), IN(3), out_m, coefs, c0,
                                                     IN(80), IN(81), IN(82), out_alpha, out_spec);
  #undef IN
  #undef ALLOCF
  #undef ALLOCH
}
